// GENIACore_12438225289545
// MI455X (gfx1250) — compile-verified, matrix operand order checked by reading, not run on hardware
//
#include <hip/hip_runtime.h>
#include <hip/hip_bf16.h>
#include <math.h>

#define DMODEL 1024
#define HQKDIM 1024
#define FFDIM  4096
#define VOCABSZ 32000
#define NCAT 75
#define UNK_IDX 1
#define EPSV 1e-5f
#define BATCH 8
#define SEQ 2048
#define NTOK (BATCH*SEQ)

typedef __attribute__((ext_vector_type(16))) _Float16 v16h;
typedef __attribute__((ext_vector_type(8)))  _Float16 v8h;
typedef __attribute__((ext_vector_type(8)))  float    v8f;

// ---- CDNA5 async global->LDS copy (16B per lane), ASYNCcnt-tracked ----
// Inline asm (VFLAT syntax: vdst = LDS byte offset VGPR, vaddr = 64-bit global
// address). The clang builtin exists but takes LangAS-qualified vector-pointer
// params that aren't portably spellable from HIP generic pointers.
__device__ __forceinline__ void g2l_async_b128(const void* g, void* l) {
  unsigned loff = (unsigned)(size_t)l;   // low 32 bits of flat addr = LDS offset
  asm volatile("global_load_async_to_lds_b128 %0, %1, off"
               :: "v"(loff), "v"(g) : "memory");
}

__device__ __forceinline__ void wait_async0() {
#if __has_builtin(__builtin_amdgcn_s_wait_asynccnt)
  __builtin_amdgcn_s_wait_asynccnt(0);
#else
  asm volatile("s_wait_asynccnt 0x0" ::: "memory");
#endif
}

// ------------------------------------------------------------------
// Embedding lookup + sinusoidal positional encoding -> x (f32 and f16)
// ------------------------------------------------------------------
__global__ void k_embed(const int* __restrict__ idx, const float* __restrict__ emb,
                        float* __restrict__ xf, _Float16* __restrict__ xh) {
  int tok = blockIdx.x;
  int pos = tok % SEQ;
  int id  = idx[tok];
  if (id >= VOCABSZ || id < 0) id = UNK_IDX;
  const float* erow = emb + (size_t)id * DMODEL;
  for (int d = threadIdx.x; d < DMODEL; d += blockDim.x) {
    int i2 = d >> 1;
    float denom = powf(10000.0f, (2.0f * (float)i2) / (float)DMODEL);
    float ang = (float)pos / denom;
    float pe = (d & 1) ? cosf(ang) : sinf(ang);
    float v = erow[d] + pe;
    size_t o = (size_t)tok * DMODEL + d;
    xf[o] = v;
    xh[o] = (_Float16)v;
  }
}

// ------------------------------------------------------------------
// Weight transpose + f32->f16 convert:  in[K,N] (f32) -> out[N,K] (f16)
// ------------------------------------------------------------------
__global__ void k_w_t(const float* __restrict__ in, _Float16* __restrict__ out,
                      int K, int N) {
  size_t total = (size_t)K * N;
  for (size_t t = (size_t)blockIdx.x * blockDim.x + threadIdx.x; t < total;
       t += (size_t)gridDim.x * blockDim.x) {
    size_t k = t / N, n = t % N;
    out[n * (size_t)K + k] = (_Float16)in[t];
  }
}

// ------------------------------------------------------------------
// WMMA GEMM, double-buffered LDS with async global->LDS staging.
//   D = scale * A[M,K] * op(B) + bias  (+relu), outputs f32 (C) and/or f16 (Ch)
//   TRANSB=1 : B is [N,K] row-major (NT; async-copy both tiles)
//   TRANSB=0 : B is [K,N] row-major (NN; B transposed through VGPRs)
// 128x128x32 tiles, 256 threads = 8 wave32 (2x4), wave tile 64x32 ->
// 4x2 v_wmma_f32_16x16x32_f16 accumulators. Dims multiples of tiles.
// ------------------------------------------------------------------
#define BM 128
#define BN 128
#define BKS 32
#define LDK 40   // padded LDS row stride in halves (80B; 16B-aligned chunks)

template<int TRANSB>
__global__ __launch_bounds__(256)
void k_gemm(const _Float16* __restrict__ A, const _Float16* __restrict__ B,
            float* __restrict__ C, _Float16* __restrict__ Ch,
            const float* __restrict__ bias,
            int M, int N, int K, float scale, int relu,
            long long sA, long long sB, long long sC) {
  __shared__ _Float16 As[2 * BM * LDK];
  __shared__ _Float16 Bs[2 * BN * LDK];

  const int tid   = threadIdx.x;
  const int lane  = tid & 31;
  const int wave  = tid >> 5;
  const int wm    = wave & 1;      // 64 rows of M per wave row
  const int wn    = wave >> 1;     // 32 cols of N per wave col
  const int lhalf = lane >> 4;
  const int l16   = lane & 15;
  const int k0    = lhalf * 8;     // per-lane K chunk base (ISA 16-bit A 16x32)

  const int bn0 = blockIdx.x * BN;
  const int bm0 = blockIdx.y * BM;
  A += (long long)blockIdx.z * sA;
  B += (long long)blockIdx.z * sB;

  auto stage = [&](int kk, int buf) {
    _Float16* Asb = As + buf * (BM * LDK);
    _Float16* Bsb = Bs + buf * (BN * LDK);
    #pragma unroll
    for (int c = tid; c < (BM * BKS) / 8; c += 256) {
      int r = c >> 2, col = (c & 3) * 8;
      g2l_async_b128(A + (size_t)(bm0 + r) * K + kk + col, &Asb[r * LDK + col]);
    }
    if constexpr (TRANSB) {
      #pragma unroll
      for (int c = tid; c < (BN * BKS) / 8; c += 256) {
        int r = c >> 2, col = (c & 3) * 8;
        g2l_async_b128(B + (size_t)(bn0 + r) * K + kk + col, &Bsb[r * LDK + col]);
      }
    } else {
      #pragma unroll
      for (int c = tid; c < (BKS * BN) / 8; c += 256) {
        int kr = c >> 4, nc = (c & 15) * 8;
        v8h v = *(const v8h*)(B + (size_t)(kk + kr) * N + bn0 + nc);
        #pragma unroll
        for (int j = 0; j < 8; j++)
          Bsb[(nc + j) * LDK + kr] = v[j];
      }
    }
  };

  v8f acc[4][2] = {};

  stage(0, 0);
  wait_async0();
  __syncthreads();

  int cur = 0;
  for (int kk = 0; kk < K; kk += BKS) {
    if (kk + BKS < K) stage(kk + BKS, cur ^ 1);

    const _Float16* Asb = As + cur * (BM * LDK);
    const _Float16* Bsb = Bs + cur * (BN * LDK);
    v16h af[4], bf[2];
    #pragma unroll
    for (int mi = 0; mi < 4; mi++) {
      const _Float16* p = &Asb[(wm * 64 + mi * 16 + l16) * LDK];
      v8h lo = *(const v8h*)(p + k0);
      v8h hi = *(const v8h*)(p + k0 + 16);
      af[mi] = __builtin_shufflevector(lo, hi, 0,1,2,3,4,5,6,7,8,9,10,11,12,13,14,15);
    }
    #pragma unroll
    for (int ni = 0; ni < 2; ni++) {
      const _Float16* p = &Bsb[(wn * 32 + ni * 16 + l16) * LDK];
      v8h lo = *(const v8h*)(p + k0);
      v8h hi = *(const v8h*)(p + k0 + 16);
      bf[ni] = __builtin_shufflevector(lo, hi, 0,1,2,3,4,5,6,7,8,9,10,11,12,13,14,15);
    }
    #pragma unroll
    for (int mi = 0; mi < 4; mi++)
      #pragma unroll
      for (int ni = 0; ni < 2; ni++)
        acc[mi][ni] = __builtin_amdgcn_wmma_f32_16x16x32_f16(
            false, af[mi], false, bf[ni], (short)0, acc[mi][ni], false, false);

    wait_async0();   // next tile's async copies (this wave) complete
    __syncthreads(); // all waves done reading cur / writing cur^1
    cur ^= 1;
  }

  // ---- epilogue: scale, bias, relu; optional f32 and/or f16 stores ----
  #pragma unroll
  for (int mi = 0; mi < 4; mi++) {
    #pragma unroll
    for (int ni = 0; ni < 2; ni++) {
      int coln = bn0 + wn * 32 + ni * 16 + l16;
      float bval = bias ? bias[coln] : 0.0f;
      #pragma unroll
      for (int i = 0; i < 8; i++) {
        int row = bm0 + wm * 64 + mi * 16 + lhalf * 8 + i;
        float v = acc[mi][ni][i] * scale + bval;
        if (relu) v = fmaxf(v, 0.0f);
        size_t o = (size_t)row * N + coln;
        if (C)  C[(size_t)blockIdx.z * sC + o]  = v;
        if (Ch) Ch[(size_t)blockIdx.z * sC + o] = (_Float16)v;
      }
    }
  }
}

// ------------------------------------------------------------------
// Fused residual + LayerNorm over DMODEL: h = LN(x+u)*g+b -> f32? & f16
// ------------------------------------------------------------------
__global__ __launch_bounds__(256)
void k_res_ln(const float* __restrict__ X, const float* __restrict__ U,
              const float* __restrict__ g, const float* __restrict__ b,
              float* __restrict__ Hf, _Float16* __restrict__ Hh) {
  __shared__ float s1[256], s2[256];
  int row = blockIdx.x;
  float loc[4];
  float s = 0.f, q = 0.f;
  #pragma unroll
  for (int j = 0; j < 4; j++) {
    int d = threadIdx.x + j * 256;
    float v = X[(size_t)row * DMODEL + d] + U[(size_t)row * DMODEL + d];
    loc[j] = v; s += v; q += v * v;
  }
  s1[threadIdx.x] = s; s2[threadIdx.x] = q;
  __syncthreads();
  for (int off = 128; off > 0; off >>= 1) {
    if (threadIdx.x < off) {
      s1[threadIdx.x] += s1[threadIdx.x + off];
      s2[threadIdx.x] += s2[threadIdx.x + off];
    }
    __syncthreads();
  }
  float mu  = s1[0] / (float)DMODEL;
  float var = s2[0] / (float)DMODEL - mu * mu;
  float rstd = rsqrtf(var + EPSV);
  #pragma unroll
  for (int j = 0; j < 4; j++) {
    int d = threadIdx.x + j * 256;
    float h = (loc[j] - mu) * rstd * g[d] + b[d];
    if (Hf) Hf[(size_t)row * DMODEL + d] = h;
    Hh[(size_t)row * DMODEL + d] = (_Float16)h;
  }
}

// ------------------------------------------------------------------
// Head: logits = r3 @ W4 + b4 (K=512,N=75), LN over 75, sigmoid -> out f32
// ------------------------------------------------------------------
__global__ __launch_bounds__(128)
void k_head(const float* __restrict__ R3, const float* __restrict__ W4,
            const float* __restrict__ b4, const float* __restrict__ g3,
            const float* __restrict__ be3, float* __restrict__ out) {
  __shared__ float r[DMODEL / 2];
  __shared__ float lg[NCAT];
  __shared__ float r1[128], r2[128];
  int tok = blockIdx.x;
  for (int i = threadIdx.x; i < DMODEL / 2; i += 128)
    r[i] = R3[(size_t)tok * (DMODEL / 2) + i];
  __syncthreads();
  if (threadIdx.x < NCAT) {
    float acc = b4[threadIdx.x];
    for (int k = 0; k < DMODEL / 2; k++)
      acc += r[k] * W4[k * NCAT + threadIdx.x];
    lg[threadIdx.x] = acc;
  }
  __syncthreads();
  float v = (threadIdx.x < NCAT) ? lg[threadIdx.x] : 0.0f;
  r1[threadIdx.x] = v; r2[threadIdx.x] = v * v;
  __syncthreads();
  for (int off = 64; off > 0; off >>= 1) {
    if (threadIdx.x < off) {
      r1[threadIdx.x] += r1[threadIdx.x + off];
      r2[threadIdx.x] += r2[threadIdx.x + off];
    }
    __syncthreads();
  }
  float mu  = r1[0] / (float)NCAT;
  float var = r2[0] / (float)NCAT - mu * mu;
  float rstd = rsqrtf(var + EPSV);
  if (threadIdx.x < NCAT) {
    float y = (lg[threadIdx.x] - mu) * rstd * g3[threadIdx.x] + be3[threadIdx.x];
    out[(size_t)tok * NCAT + threadIdx.x] = 1.0f / (1.0f + expf(-y));
  }
}

// ------------------------------------------------------------------
extern "C" void kernel_launch(void* const* d_in, const int* in_sizes, int n_in,
                              void* d_out, int out_size, void* d_ws, size_t ws_size,
                              hipStream_t stream) {
  (void)in_sizes; (void)n_in; (void)out_size; (void)ws_size;

  const int*   idx = (const int*)  d_in[0];
  const float* emb = (const float*)d_in[1];
  const float* Wq  = (const float*)d_in[2];
  const float* bq  = (const float*)d_in[3];
  const float* Wk  = (const float*)d_in[4];
  const float* bk  = (const float*)d_in[5];
  const float* Wv  = (const float*)d_in[6];
  const float* bv  = (const float*)d_in[7];
  const float* Wu  = (const float*)d_in[8];
  const float* bu  = (const float*)d_in[9];
  const float* g1  = (const float*)d_in[10];
  const float* be1 = (const float*)d_in[11];
  const float* W1  = (const float*)d_in[12];
  const float* b1  = (const float*)d_in[13];
  const float* W2  = (const float*)d_in[14];
  const float* b2  = (const float*)d_in[15];
  const float* g2  = (const float*)d_in[16];
  const float* be2 = (const float*)d_in[17];
  const float* W3  = (const float*)d_in[18];
  const float* b3  = (const float*)d_in[19];
  const float* W4  = (const float*)d_in[20];
  const float* b4  = (const float*)d_in[21];
  const float* g3  = (const float*)d_in[22];
  const float* be3 = (const float*)d_in[23];

  char* ws = (char*)d_ws;
  size_t off = 0;
  auto alloc = [&](size_t bytes) -> char* {
    char* p = ws + off;
    off = (off + bytes + 255) & ~(size_t)255;
    return p;
  };

  float*    x_f32   = (float*)   alloc((size_t)NTOK * DMODEL * 4);
  _Float16* x_f16   = (_Float16*)alloc((size_t)NTOK * DMODEL * 2);
  _Float16* q_f16   = (_Float16*)alloc((size_t)NTOK * HQKDIM * 2);
  _Float16* kk_f16  = (_Float16*)alloc((size_t)NTOK * HQKDIM * 2);
  _Float16* v_f16   = (_Float16*)alloc((size_t)NTOK * HQKDIM * 2);
  _Float16* sc_f16  = (_Float16*)alloc((size_t)BATCH * SEQ * SEQ * 2);
  _Float16* at_f16  = (_Float16*)alloc((size_t)NTOK * HQKDIM * 2);
  float*    h1_f32  = (float*)   alloc((size_t)NTOK * DMODEL * 4);
  _Float16* h1_f16  = (_Float16*)alloc((size_t)NTOK * DMODEL * 2);
  _Float16* ff1_f16 = (_Float16*)alloc((size_t)NTOK * FFDIM * 2);
  _Float16* h2_f16  = (_Float16*)alloc((size_t)NTOK * DMODEL * 2);
  _Float16* wqT = (_Float16*)alloc((size_t)DMODEL * HQKDIM * 2);
  _Float16* wkT = (_Float16*)alloc((size_t)DMODEL * HQKDIM * 2);
  _Float16* wvT = (_Float16*)alloc((size_t)DMODEL * HQKDIM * 2);
  _Float16* wuT = (_Float16*)alloc((size_t)HQKDIM * DMODEL * 2);
  _Float16* w1T = (_Float16*)alloc((size_t)DMODEL * FFDIM * 2);
  _Float16* w2T = (_Float16*)alloc((size_t)FFDIM * DMODEL * 2);
  _Float16* w3T = (_Float16*)alloc((size_t)DMODEL * (DMODEL / 2) * 2);
  float*    tmp = (float*)alloc((size_t)NTOK * DMODEL * 4);  // f32 GEMM out (LN/head input)

  // ---- prep ----
  k_embed<<<NTOK, 256, 0, stream>>>(idx, emb, x_f32, x_f16);
  k_w_t<<<2048, 256, 0, stream>>>(Wq, wqT, DMODEL, HQKDIM);
  k_w_t<<<2048, 256, 0, stream>>>(Wk, wkT, DMODEL, HQKDIM);
  k_w_t<<<2048, 256, 0, stream>>>(Wv, wvT, DMODEL, HQKDIM);
  k_w_t<<<2048, 256, 0, stream>>>(Wu, wuT, HQKDIM, DMODEL);
  k_w_t<<<4096, 256, 0, stream>>>(W1, w1T, DMODEL, FFDIM);
  k_w_t<<<4096, 256, 0, stream>>>(W2, w2T, FFDIM, DMODEL);
  k_w_t<<<1024, 256, 0, stream>>>(W3, w3T, DMODEL, DMODEL / 2);

  auto gemmNT = [&](const _Float16* A, const _Float16* B, float* C, _Float16* Ch,
                    const float* bias, int M, int N, int K, float scl, int relu,
                    int bat, long long sA, long long sB, long long sC) {
    dim3 grid(N / BN, M / BM, bat);
    k_gemm<1><<<grid, 256, 0, stream>>>(A, B, C, Ch, bias, M, N, K, scl, relu, sA, sB, sC);
  };
  auto gemmNN = [&](const _Float16* A, const _Float16* B, float* C, _Float16* Ch,
                    const float* bias, int M, int N, int K, float scl, int relu,
                    int bat, long long sA, long long sB, long long sC) {
    dim3 grid(N / BN, M / BM, bat);
    k_gemm<0><<<grid, 256, 0, stream>>>(A, B, C, Ch, bias, M, N, K, scl, relu, sA, sB, sC);
  };

  // ---- QKV projections (f16 out, fused) ----
  gemmNT(x_f16, wqT, nullptr, q_f16,  bq, NTOK, HQKDIM, DMODEL, 1.f, 0, 1, 0, 0, 0);
  gemmNT(x_f16, wkT, nullptr, kk_f16, bk, NTOK, HQKDIM, DMODEL, 1.f, 0, 1, 0, 0, 0);
  gemmNT(x_f16, wvT, nullptr, v_f16,  bv, NTOK, HQKDIM, DMODEL, 1.f, 0, 1, 0, 0, 0);

  // ---- scores = Q K^T / sqrt(64)  (batched NT, f16 out) ----
  gemmNT(q_f16, kk_f16, nullptr, sc_f16, nullptr, SEQ, SEQ, HQKDIM, 0.125f, 0, BATCH,
         (long long)SEQ * HQKDIM, (long long)SEQ * HQKDIM, (long long)SEQ * SEQ);

  // ---- attn = scores @ V  (batched NN, f16 out) ----
  gemmNN(sc_f16, v_f16, nullptr, at_f16, nullptr, SEQ, HQKDIM, SEQ, 1.f, 0, BATCH,
         (long long)SEQ * SEQ, (long long)SEQ * HQKDIM, (long long)SEQ * HQKDIM);

  // ---- u = attn @ Wu + bu (f32) ; h1 = LN(x+u) ----
  gemmNT(at_f16, wuT, tmp, nullptr, bu, NTOK, DMODEL, HQKDIM, 1.f, 0, 1, 0, 0, 0);
  k_res_ln<<<NTOK, 256, 0, stream>>>(x_f32, tmp, g1, be1, h1_f32, h1_f16);

  // ---- FFN ----
  gemmNT(h1_f16, w1T, nullptr, ff1_f16, b1, NTOK, FFDIM, DMODEL, 1.f, 1, 1, 0, 0, 0);
  gemmNT(ff1_f16, w2T, tmp, nullptr, b2, NTOK, DMODEL, FFDIM, 1.f, 0, 1, 0, 0, 0);
  k_res_ln<<<NTOK, 256, 0, stream>>>(h1_f32, tmp, g2, be2, nullptr, h2_f16);

  // ---- head: r3 = relu(h2 @ W3 + b3) (f32) ; logits/LN/sigmoid ----
  gemmNT(h2_f16, w3T, tmp, nullptr, b3, NTOK, DMODEL / 2, DMODEL, 1.f, 1, 1, 0, 0, 0);
  k_head<<<NTOK, 128, 0, stream>>>(tmp, W4, b4, g3, be3, (float*)d_out);
}
